// Attention_9354438771128
// MI455X (gfx1250) — matrix operand order repaired and checked by reading, not compiled, not run on hardware
//
#include <hip/hip_runtime.h>
#include <hip/hip_bf16.h>
#include <stdint.h>

#define DIM 4096
#define N_HEADS 32
#define HEAD_DIM 128
#define N_KV_HEADS 8
#define NB 8
#define NS 1024
#define NT (NB * NS)
#define QSCALE 0.08838834764831845f   // 128^-0.5

typedef __attribute__((ext_vector_type(16))) __bf16 v16bf;
typedef __attribute__((ext_vector_type(8)))  float  v8f;
typedef __attribute__((ext_vector_type(4)))  unsigned int v4u;
typedef __attribute__((ext_vector_type(8)))  int v8i;
typedef __attribute__((ext_vector_type(4)))  int v4i;

union Frag {
  v16bf v;
  uint4 q[2];
  unsigned short u[16];
};

__device__ __forceinline__ unsigned short f2bf(float f) {
  unsigned int u = __builtin_bit_cast(unsigned int, f);
  u += 0x7FFFu + ((u >> 16) & 1u);           // round-to-nearest-even
  return (unsigned short)(u >> 16);
}

__device__ __forceinline__ v8f wmma_bf16(const Frag& a, const Frag& b, v8f c) {
  return __builtin_amdgcn_wmma_f32_16x16x32_bf16(false, a.v, false, b.v,
                                                 (short)0, c, false, false);
}

// ---------------- fp32 -> bf16 flat convert ----------------
__global__ void cvt_flat(const float* __restrict__ in,
                         unsigned short* __restrict__ out, int n) {
  int i = blockIdx.x * blockDim.x + threadIdx.x;
  if (i < n) out[i] = f2bf(in[i]);
}

// ---------- fp32 [K,N] -> bf16 transposed [N,K] (weights) ----------
__global__ void cvt_t(const float* __restrict__ in,
                      unsigned short* __restrict__ out, int K, int N) {
  long long i = (long long)blockIdx.x * blockDim.x + threadIdx.x;
  long long total = (long long)K * N;
  if (i < total) {
    int k = (int)(i / N);
    int n = (int)(i % N);
    out[(size_t)n * K + k] = f2bf(in[i]);
  }
}

// ---------------- RoPE + repack [T,H*128] -> bf16 [B,H,S,128] ----------------
__global__ void rope_pack(const float* __restrict__ in,
                          const float* __restrict__ cosb,
                          const float* __restrict__ sinb,
                          unsigned short* __restrict__ out,
                          int H, float scale) {
  int i = blockIdx.x * blockDim.x + threadIdx.x;   // over T*H*64
  int pair = i & 63;
  int h = (i >> 6) % H;
  int t = i / (64 * H);
  if (t >= NT) return;
  float c = cosb[t * 64 + pair];
  float s = sinb[t * 64 + pair];
  const float* p = in + (size_t)t * H * HEAD_DIM + h * HEAD_DIM + 2 * pair;
  float x1 = p[0], x2 = p[1];
  float o1 = (x1 * c - x2 * s) * scale;
  float o2 = (x1 * s + x2 * c) * scale;
  int b = t / NS, srow = t % NS;
  size_t o = (((size_t)b * H + h) * NS + srow) * HEAD_DIM + 2 * pair;
  out[o]     = f2bf(o1);
  out[o + 1] = f2bf(o2);
}

// ----- V repack: fp32 [T,KV*128] -> bf16 transposed [B,KV,128,S] -----
__global__ void v_pack(const float* __restrict__ in,
                       unsigned short* __restrict__ out) {
  int i = blockIdx.x * blockDim.x + threadIdx.x;   // over T*KV*128
  int d = i & (HEAD_DIM - 1);
  int h = (i >> 7) % N_KV_HEADS;
  int t = i / (HEAD_DIM * N_KV_HEADS);
  if (t >= NT) return;
  int b = t / NS, srow = t % NS;
  out[(((size_t)b * N_KV_HEADS + h) * HEAD_DIM + d) * NS + srow] = f2bf(in[i]);
}

// ------------- bf16 WMMA GEMM with TDM-staged A tiles -------------
// C[M,N] = A[M,K] * Bt[N,K]^T.  Block = 256 thr (8 waves) -> 64x128 C tile.
// Waves: wm in {0,1} (32 rows each), wn in {0..3} (32 cols each);
// each wave computes 32x32 via 2x2 register blocking (4 wmma / k-step).
// A 64x32 bf16 block-tile is DMA'd into LDS by the Tensor Data Mover,
// double-buffered so the next tile's DMA overlaps current wmma work.
__global__ void gemm_bf16(const unsigned short* __restrict__ A,
                          const unsigned short* __restrict__ Bt,
                          float* __restrict__ C,
                          int M, int N, int K) {
  __shared__ unsigned short ldsA[2][64 * 32];   // 2 x 4KB

  int lane = threadIdx.x & 31;
  int wave = threadIdx.x >> 5;                  // 0..7
  int half = lane >> 4;
  int nl   = lane & 15;
  int wm = wave & 1;
  int wn = wave >> 1;
  int mblk = blockIdx.x * 64;
  int n0   = blockIdx.y * 128 + wn * 32;

  const unsigned short* brow0 = Bt + (size_t)(n0 + nl) * K;
  const unsigned short* brow1 = Bt + (size_t)(n0 + 16 + nl) * K;

  unsigned ldsOff0 = (unsigned)(uintptr_t)&ldsA[0][0];
  unsigned ldsOff1 = (unsigned)(uintptr_t)&ldsA[1][0];

  // Issue one TDM descriptor: 2D tile 32(k) x 64(rows) of 2-byte elems.
  auto issue_tdm = [&](int buf, int kk) {
    unsigned long long ga =
        (unsigned long long)(uintptr_t)(A + (size_t)mblk * K + kk);
    unsigned td0 = (unsigned)K;       // tensor_dim0 (elements, contiguous k)
    unsigned td1 = (unsigned)M;       // tensor_dim1 (rows)
    v4u g0;
    g0[0] = 1u;                                       // count=1, user mode
    g0[1] = buf ? ldsOff1 : ldsOff0;                  // lds_addr
    g0[2] = (unsigned)(ga & 0xFFFFFFFFull);           // global_addr[31:0]
    g0[3] = (unsigned)((ga >> 32) & 0x01FFFFFFull)    // global_addr[56:32]
            | (2u << 30);                             // type = 2 (image)
    v8i g1;
    g1[0] = (int)(1u << 16);                          // data_size=1 -> 2 bytes
    g1[1] = (int)((td0 & 0xFFFFu) << 16);             // tensor_dim0[15:0]
    g1[2] = (int)((td0 >> 16) | ((td1 & 0xFFFFu) << 16));
    g1[3] = (int)((td1 >> 16) | (32u << 16));         // tile_dim0 = 32
    g1[4] = (int)64;                                  // tile_dim1 = 64
    g1[5] = (int)td0;                                 // dim0_stride[31:0] = K
    g1[6] = 0;                                        // dim0_stride[47:32]
    g1[7] = 0;
    v4i z4 = (v4i)0;
    v8i z8 = (v8i)0;
    __builtin_amdgcn_tensor_load_to_lds(g0, g1, z4, z4, z8, 0);
  };

  v8f acc00 = {}, acc01 = {}, acc10 = {}, acc11 = {};

  int nk = K >> 5;
  if (wave == 0) issue_tdm(0, 0);

  for (int t = 0; t < nk; ++t) {
    int kk  = t << 5;
    int buf = t & 1;
    if (wave == 0) {
      if (t + 1 < nk) {
        issue_tdm(buf ^ 1, kk + 32);                  // overlap next DMA
        __builtin_amdgcn_s_wait_tensorcnt(1);         // current tile ready
      } else {
        __builtin_amdgcn_s_wait_tensorcnt(0);
      }
    }
    __syncthreads();                                  // publish LDS tile

    const unsigned short* la = &ldsA[buf][0];
    Frag a0, a1, b0, b1;
    a0.q[0] = *(const uint4*)(la + (wm * 32 + nl) * 32 + half * 16);
    a0.q[1] = *(const uint4*)(la + (wm * 32 + nl) * 32 + half * 16 + 8);
    a1.q[0] = *(const uint4*)(la + (wm * 32 + 16 + nl) * 32 + half * 16);
    a1.q[1] = *(const uint4*)(la + (wm * 32 + 16 + nl) * 32 + half * 16 + 8);
    const uint4* pb0 = (const uint4*)(brow0 + kk + half * 16);
    const uint4* pb1 = (const uint4*)(brow1 + kk + half * 16);
    b0.q[0] = pb0[0]; b0.q[1] = pb0[1];
    b1.q[0] = pb1[0]; b1.q[1] = pb1[1];
    if (kk + 32 < K) {
      __builtin_prefetch(brow0 + kk + 32, 0, 1);
      __builtin_prefetch(brow1 + kk + 32, 0, 1);
    }

    acc00 = wmma_bf16(a0, b0, acc00);
    acc01 = wmma_bf16(a0, b1, acc01);
    acc10 = wmma_bf16(a1, b0, acc10);
    acc11 = wmma_bf16(a1, b1, acc11);

    __syncthreads();                                  // reads done before overwrite
  }

  // C layout per tile: m = r + 8*half, n = lane&15
  int m0 = mblk + wm * 32;
  float* c00 = C + (size_t)(m0 + 8 * half) * N + n0 + nl;
  float* c10 = C + (size_t)(m0 + 16 + 8 * half) * N + n0 + nl;
#pragma unroll
  for (int r = 0; r < 8; ++r) {
    c00[(size_t)r * N]      = acc00[r];
    c00[(size_t)r * N + 16] = acc01[r];
    c10[(size_t)r * N]      = acc10[r];
    c10[(size_t)r * N + 16] = acc11[r];
  }
}

// ---------------- flash attention (online softmax, bf16 WMMA) ----------------
// Q [B,H,S,128] (pre-scaled), K [B,KV,S,128], Vt [B,KV,128,S]; O bf16 [T, H*128]
__global__ void flash_attn(const unsigned short* __restrict__ Q,
                           const unsigned short* __restrict__ Kb,
                           const unsigned short* __restrict__ Vt,
                           unsigned short* __restrict__ O) {
  __shared__ unsigned short ldsP[4][16 * 32];   // per-wave 16x32 P tile

  int lane = threadIdx.x & 31;
  int wave = threadIdx.x >> 5;                  // 0..3
  int half = lane >> 4;
  int nl   = lane & 15;

  int bid = blockIdx.x;
  int qt  = bid & 15;                           // S/64 = 16 q-tiles
  int h   = (bid >> 4) & 31;
  int b   = bid >> 9;
  int kvh = h >> 2;                             // REPEATS = 4
  int q0  = qt * 64 + wave * 16;

  const unsigned short* qbase = Q  + (((size_t)b * N_HEADS + h) * NS + q0 + nl) * HEAD_DIM;
  const unsigned short* kbase = Kb + ((size_t)b * N_KV_HEADS + kvh) * NS * HEAD_DIM;
  const unsigned short* vbase = Vt + ((size_t)b * N_KV_HEADS + kvh) * HEAD_DIM * NS;

  Frag qa[4];
#pragma unroll
  for (int kc = 0; kc < 4; ++kc) {
    qa[kc].q[0] = *reinterpret_cast<const uint4*>(qbase + kc * 32 + half * 16);
    qa[kc].q[1] = *reinterpret_cast<const uint4*>(qbase + kc * 32 + half * 16 + 8);
  }

  float mrow[8], lrow[8];
  v8f o[8];
#pragma unroll
  for (int r = 0; r < 8; ++r) { mrow[r] = -1e30f; lrow[r] = 0.f; }
#pragma unroll
  for (int d = 0; d < 8; ++d) o[d] = (v8f){};

  for (int k0 = 0; k0 < NS; k0 += 32) {
    v8f s0 = {}, s1 = {};
#pragma unroll
    for (int kc = 0; kc < 4; ++kc) {
      Frag kf0, kf1;
      const unsigned short* kp0 = kbase + (size_t)(k0 + nl) * HEAD_DIM + kc * 32 + half * 16;
      const unsigned short* kp1 = kbase + (size_t)(k0 + 16 + nl) * HEAD_DIM + kc * 32 + half * 16;
      kf0.q[0] = ((const uint4*)kp0)[0]; kf0.q[1] = ((const uint4*)kp0)[1];
      kf1.q[0] = ((const uint4*)kp1)[0]; kf1.q[1] = ((const uint4*)kp1)[1];
      s0 = wmma_bf16(qa[kc], kf0, s0);
      s1 = wmma_bf16(qa[kc], kf1, s1);
    }

    float p0[8], p1[8];
#pragma unroll
    for (int r = 0; r < 8; ++r) {
      float mx = fmaxf(s0[r], s1[r]);
#pragma unroll
      for (int off = 8; off >= 1; off >>= 1)
        mx = fmaxf(mx, __shfl_xor(mx, off, 32));
      float mn   = fmaxf(mrow[r], mx);
      float corr = __expf(mrow[r] - mn);
      float e0   = __expf(s0[r] - mn);
      float e1   = __expf(s1[r] - mn);
      float rs   = e0 + e1;
#pragma unroll
      for (int off = 8; off >= 1; off >>= 1)
        rs += __shfl_xor(rs, off, 32);
      lrow[r] = lrow[r] * corr + rs;
      mrow[r] = mn;
      p0[r] = e0;
      p1[r] = e1;
#pragma unroll
      for (int d = 0; d < 8; ++d) o[d][r] = o[d][r] * corr;
    }

    // re-layout P (C layout -> A fragment) via per-wave LDS bounce
#pragma unroll
    for (int r = 0; r < 8; ++r) {
      int m = r + 8 * half;
      ldsP[wave][m * 32 + nl]      = f2bf(p0[r]);
      ldsP[wave][m * 32 + 16 + nl] = f2bf(p1[r]);
    }
    __builtin_amdgcn_wave_barrier();   // per-wave LDS ops are in-order (DScnt)
    Frag pf;
    pf.q[0] = *reinterpret_cast<const uint4*>(&ldsP[wave][nl * 32 + half * 16]);
    pf.q[1] = *reinterpret_cast<const uint4*>(&ldsP[wave][nl * 32 + half * 16 + 8]);

#pragma unroll
    for (int dt = 0; dt < 8; ++dt) {
      Frag vf;
      const unsigned short* vp = vbase + (size_t)(dt * 16 + nl) * NS + k0 + half * 16;
      vf.q[0] = ((const uint4*)vp)[0];
      vf.q[1] = ((const uint4*)vp)[1];
      o[dt] = wmma_bf16(pf, vf, o[dt]);
    }
  }

  int token0 = b * NS + q0 + 8 * half;
#pragma unroll
  for (int r = 0; r < 8; ++r) {
    float inv = 1.0f / lrow[r];
    size_t orow = (size_t)(token0 + r) * (N_HEADS * HEAD_DIM) + (size_t)h * HEAD_DIM;
#pragma unroll
    for (int dt = 0; dt < 8; ++dt)
      O[orow + dt * 16 + nl] = f2bf(o[dt][r] * inv);
  }
}

// ---------------------------------------------------------------------------
extern "C" void kernel_launch(void* const* d_in, const int* in_sizes, int n_in,
                              void* d_out, int out_size, void* d_ws, size_t ws_size,
                              hipStream_t stream) {
  const float* x  = (const float*)d_in[0];
  const float* fc = (const float*)d_in[1];
  const float* fs = (const float*)d_in[2];
  const float* wq = (const float*)d_in[3];
  const float* wk = (const float*)d_in[4];
  const float* wv = (const float*)d_in[5];
  const float* wo = (const float*)d_in[6];
  float* out = (float*)d_out;

  char* ws = (char*)d_ws;
  size_t off = 0;
  auto alloc = [&](size_t bytes) -> char* {
    char* p = ws + off;
    off += (bytes + 255) & ~(size_t)255;
    return p;
  };
  unsigned short* xb  = (unsigned short*)alloc((size_t)NT * DIM * 2);
  unsigned short* wqb = (unsigned short*)alloc((size_t)DIM * N_HEADS * HEAD_DIM * 2);
  unsigned short* wkb = (unsigned short*)alloc((size_t)DIM * N_KV_HEADS * HEAD_DIM * 2);
  unsigned short* wvb = (unsigned short*)alloc((size_t)DIM * N_KV_HEADS * HEAD_DIM * 2);
  unsigned short* wob = (unsigned short*)alloc((size_t)N_HEADS * HEAD_DIM * DIM * 2);
  float* qf = (float*)alloc((size_t)NT * N_HEADS * HEAD_DIM * 4);
  float* kf = (float*)alloc((size_t)NT * N_KV_HEADS * HEAD_DIM * 4);
  float* vf = (float*)alloc((size_t)NT * N_KV_HEADS * HEAD_DIM * 4);
  unsigned short* qr = (unsigned short*)alloc((size_t)NT * N_HEADS * HEAD_DIM * 2);
  unsigned short* kr = (unsigned short*)alloc((size_t)NT * N_KV_HEADS * HEAD_DIM * 2);
  unsigned short* vr = (unsigned short*)alloc((size_t)NT * N_KV_HEADS * HEAD_DIM * 2);
  // attn output reuses the (dead by then) qf fp32 region: 64MB <= 128MB
  unsigned short* attn = (unsigned short*)qf;

  const int thr = 256;
  const int NQ = N_HEADS * HEAD_DIM;     // 4096
  const int NKV = N_KV_HEADS * HEAD_DIM; // 1024

  cvt_flat<<<(NT * DIM + thr - 1) / thr, thr, 0, stream>>>(x, xb, NT * DIM);
  cvt_t<<<(DIM * NQ + thr - 1) / thr, thr, 0, stream>>>(wq, wqb, DIM, NQ);
  cvt_t<<<(DIM * NKV + thr - 1) / thr, thr, 0, stream>>>(wk, wkb, DIM, NKV);
  cvt_t<<<(DIM * NKV + thr - 1) / thr, thr, 0, stream>>>(wv, wvb, DIM, NKV);
  cvt_t<<<(NQ * DIM + thr - 1) / thr, thr, 0, stream>>>(wo, wob, NQ, DIM);

  gemm_bf16<<<dim3(NT / 64, NQ / 128), 256, 0, stream>>>(xb, wqb, qf, NT, NQ, DIM);
  gemm_bf16<<<dim3(NT / 64, NKV / 128), 256, 0, stream>>>(xb, wkb, kf, NT, NKV, DIM);
  gemm_bf16<<<dim3(NT / 64, NKV / 128), 256, 0, stream>>>(xb, wvb, vf, NT, NKV, DIM);

  rope_pack<<<(NT * N_HEADS * 64) / thr, thr, 0, stream>>>(qf, fc, fs, qr, N_HEADS, QSCALE);
  rope_pack<<<(NT * N_KV_HEADS * 64) / thr, thr, 0, stream>>>(kf, fc, fs, kr, N_KV_HEADS, 1.0f);
  v_pack<<<(NT * NKV) / thr, thr, 0, stream>>>(vf, vr);

  flash_attn<<<NB * N_HEADS * (NS / 64), 128, 0, stream>>>(qr, kr, vr, attn);

  gemm_bf16<<<dim3(NT / 64, DIM / 128), 256, 0, stream>>>(attn, wob, out, NT, DIM, NQ);
}
